// GraphWavelet_RSR_79869211836445
// MI455X (gfx1250) — matrix operand order repaired and checked by reading, not compiled, not run on hardware
//
#include <hip/hip_runtime.h>

typedef __attribute__((ext_vector_type(16))) _Float16 v16h;
typedef __attribute__((ext_vector_type(8)))  float    v8f;

namespace {
constexpr int B    = 4;
constexpr int N    = 50000;
constexpr int R    = 100000;
constexpr int E    = 1600000;
constexpr int H2   = 16;
constexpr int M    = B * N;          // 200000 rows, divisible by 16
constexpr float NEG_SLOPE = 0.2f;
}

// ---------------------------------------------------------------------------
// Dense feature transform H[M x DOUT] = f16(X[M x 32]) * f16(W[32 x DOUT]),
// f32 accumulate, via V_WMMA_F32_16X16X32_F16. One wave per 16x16 output tile,
// K=32 consumed by a single WMMA (shape matches exactly).
// ---------------------------------------------------------------------------
__global__ void gw_gemm_wmma(const float* __restrict__ X,
                             const float* __restrict__ W,
                             float* __restrict__ H, int DOUT) {
  const int lane  = threadIdx.x;      // blockDim.x == 32: exactly one wave
  const int m16   = lane & 15;
  const int g     = lane >> 4;
  const int tileM = blockIdx.x * 16;
  const int tileN = blockIdx.y * 16;

  // A tile: row (tileM+m16), per-lane K layout per ISA 7.12.2 (16-bit A 16x32):
  //   halfs 0..7  -> K = g*8 + j,   halfs 8..15 -> K = 16 + g*8 + j
  const float* xr = X + (size_t)(tileM + m16) * 32;
  v16h a;
#pragma unroll
  for (int j = 0; j < 8; ++j) {
    a[j]     = (_Float16)xr[g * 8 + j];
    a[8 + j] = (_Float16)xr[16 + g * 8 + j];
  }

  // B tile: col n = tileN + (lane%16); per-lane K matches A: halfs 0..7 -> K = g*8 + j, halfs 8..15 -> K = 16 + g*8 + j
  const int n = tileN + m16;
  v16h b;
#pragma unroll
  for (int j = 0; j < 16; ++j)
    b[j] = (_Float16)W[((j >> 3) * 16 + g * 8 + (j & 7)) * DOUT + n];

  v8f c = {};
  c = __builtin_amdgcn_wmma_f32_16x16x32_f16(false, a, false, b,
                                             (short)0, c, false, false);

  // C layout: VGPR v -> row (v + 8*g), lane%16 -> col
  float* hr = H + (size_t)tileM * DOUT + tileN;
#pragma unroll
  for (int v = 0; v < 8; ++v)
    hr[(size_t)(v + 8 * g) * DOUT + m16] = c[v];
}

// ---------------------------------------------------------------------------
// COO edge scatter: dst[b, row, :] += src[b, col, :] * val (* diag[col]).
// 8 (d=32) or 4 (d=16) consecutive threads share one edge; each thread moves
// one float4 chunk for all B batches. Gathers/atomics resolve in the 192MB L2.
// ---------------------------------------------------------------------------
template <int D>
__global__ void gw_scatter(const int*   __restrict__ idx,   // (2,E): rows, cols
                           const float* __restrict__ vals,
                           const float* __restrict__ diag,  // nullptr or [col]
                           const float* __restrict__ src,   // (B, srcRows, D)
                           float*       __restrict__ dst,   // (B, dstRows, D)
                           int srcRows, int dstRows) {
  constexpr int CH  = D / 4;
  constexpr int CHS = (D == 32) ? 3 : 2;
  long long tid = (long long)blockIdx.x * blockDim.x + threadIdx.x;
  if (tid >= (long long)E * CH) return;
  const int e = (int)(tid >> CHS);
  const int c = (int)(tid & (CH - 1));

  const int row = idx[e];
  const int col = idx[E + e];
  float v = vals[e];
  if (diag) v *= diag[col];

  const int off = c << 2;
#pragma unroll
  for (int b = 0; b < B; ++b) {
    const float4 s = *(const float4*)(src + ((size_t)b * srcRows + col) * D + off);
    float* dp = dst + ((size_t)b * dstRows + row) * D + off;
    atomicAdd(dp + 0, s.x * v);
    atomicAdd(dp + 1, s.y * v);
    atomicAdd(dp + 2, s.z * v);
    atomicAdd(dp + 3, s.w * v);
  }
}

__global__ void gw_zero4(float4* __restrict__ p, long long n4) {
  long long i = (long long)blockIdx.x * blockDim.x + threadIdx.x;
  if (i < n4) p[i] = make_float4(0.f, 0.f, 0.f, 0.f);
}

__global__ void gw_leaky(float* __restrict__ p, long long n) {
  long long i = (long long)blockIdx.x * blockDim.x + threadIdx.x;
  if (i < n) {
    float v = p[i];
    p[i] = v > 0.f ? v : NEG_SLOPE * v;
  }
}

// out[b,0] = x2[b, N-2, :] . w_co + b_co ; out[b,1] = x2[b, N-1, :] . w_co2 + b_co2
__global__ void gw_readout(const float* __restrict__ x2,
                           const float* __restrict__ w_co,  const float* __restrict__ b_co,
                           const float* __restrict__ w_co2, const float* __restrict__ b_co2,
                           float* __restrict__ out) {
  const int t = threadIdx.x;
  if (t >= 2 * B) return;
  const int b     = t >> 1;
  const int which = t & 1;
  const float* row = x2 + ((size_t)b * N + (size_t)(N - 2 + which)) * H2;
  const float* w   = which ? w_co2 : w_co;
  float acc        = which ? b_co2[0] : b_co[0];
#pragma unroll
  for (int j = 0; j < H2; ++j) acc += row[j] * w[j];
  out[b * 2 + which] = acc;
}

// ---------------------------------------------------------------------------
extern "C" void kernel_launch(void* const* d_in, const int* in_sizes, int n_in,
                              void* d_out, int out_size, void* d_ws, size_t ws_size,
                              hipStream_t stream) {
  const int*   w_idx  = (const int*)  d_in[0];
  const float* w_val  = (const float*)d_in[1];
  const int*   wi_idx = (const int*)  d_in[2];
  const float* wi_val = (const float*)d_in[3];
  const float* x_feat = (const float*)d_in[4];
  const float* W1     = (const float*)d_in[5];
  const float* diag1  = (const float*)d_in[6];
  const float* W2     = (const float*)d_in[7];
  const float* diag2  = (const float*)d_in[8];
  const float* rc_w   = (const float*)d_in[9];
  const float* rc_b   = (const float*)d_in[10];
  const float* rc2_w  = (const float*)d_in[11];
  const float* rc2_b  = (const float*)d_in[12];
  float* out = (float*)d_out;

  // Workspace: h | t | x  (~102 MB total, fits comfortably in L2)
  float* h = (float*)d_ws;                 // B*N*32 = 6.4M floats
  float* t = h + (size_t)B * N * 32;       // B*R*32 = 12.8M floats
  float* x = t + (size_t)B * R * 32;       // B*N*32 = 6.4M floats

  const int mt = M / 16;                   // 12500 row tiles

  // -------------------- layer 1 (d = 32) --------------------
  gw_gemm_wmma<<<dim3(mt, 2), 32, 0, stream>>>(x_feat, W1, h, 32);

  {
    long long n4 = (long long)B * R * 32 / 4;
    gw_zero4<<<(unsigned)((n4 + 255) / 256), 256, 0, stream>>>((float4*)t, n4);
  }
  {
    long long nt = (long long)E * 8;
    gw_scatter<32><<<(unsigned)((nt + 255) / 256), 256, 0, stream>>>(
        w_idx, w_val, nullptr, h, t, N, R);
  }
  {
    long long n4 = (long long)B * N * 32 / 4;
    gw_zero4<<<(unsigned)((n4 + 255) / 256), 256, 0, stream>>>((float4*)x, n4);
  }
  {
    long long nt = (long long)E * 8;
    gw_scatter<32><<<(unsigned)((nt + 255) / 256), 256, 0, stream>>>(
        wi_idx, wi_val, diag1, t, x, R, N);
  }
  {
    long long n = (long long)B * N * 32;
    gw_leaky<<<(unsigned)((n + 255) / 256), 256, 0, stream>>>(x, n);
  }

  // -------------------- layer 2 (d = 16) --------------------
  gw_gemm_wmma<<<dim3(mt, 1), 32, 0, stream>>>(x, W2, h, 16);

  {
    long long n4 = (long long)B * R * 16 / 4;
    gw_zero4<<<(unsigned)((n4 + 255) / 256), 256, 0, stream>>>((float4*)t, n4);
  }
  {
    long long nt = (long long)E * 4;
    gw_scatter<16><<<(unsigned)((nt + 255) / 256), 256, 0, stream>>>(
        w_idx, w_val, nullptr, h, t, N, R);
  }
  {
    long long n4 = (long long)B * N * 16 / 4;   // x reused for layer-2 output
    gw_zero4<<<(unsigned)((n4 + 255) / 256), 256, 0, stream>>>((float4*)x, n4);
  }
  {
    long long nt = (long long)E * 4;
    gw_scatter<16><<<(unsigned)((nt + 255) / 256), 256, 0, stream>>>(
        wi_idx, wi_val, diag2, t, x, R, N);
  }
  {
    long long n = (long long)B * N * 16;
    gw_leaky<<<(unsigned)((n + 255) / 256), 256, 0, stream>>>(x, n);
  }

  // -------------------- readout --------------------
  gw_readout<<<1, 32, 0, stream>>>(x, rc_w, rc_b, rc2_w, rc2_b, out);
}
